// DRA_C_30966714204439
// MI455X (gfx1250) — matrix operand order repaired and checked by reading, not compiled, not run on hardware
//
#include <hip/hip_runtime.h>
#include <math.h>

// ---------------------------------------------------------------------------
// CDNA5 (gfx1250) implementation of the decoder/transformer attention block.
// wave32; WMMA f32_16x16x32_f16 for GEMMs; TDM tensor_load_to_lds for the
// strided A-tile of the 1x1-conv GEMM; LDS-staged f16 fragments elsewhere.
// ---------------------------------------------------------------------------

typedef __attribute__((ext_vector_type(16))) _Float16     v16h;
typedef __attribute__((ext_vector_type(8)))  _Float16     v8h;
typedef __attribute__((ext_vector_type(8)))  float        v8f;
typedef __attribute__((ext_vector_type(4)))  unsigned int v4u;
typedef __attribute__((ext_vector_type(8)))  int          v8i_t;
typedef __attribute__((ext_vector_type(4)))  int          v4i_t;

// Convert 16 consecutive fp32 (16B-aligned) to an f16 B/A fragment where
// element e holds k = e (contiguous K run).
__device__ __forceinline__ v16h load_f32x16_as_f16(const float* __restrict__ p)
{
    const float4 x0 = ((const float4*)p)[0];
    const float4 x1 = ((const float4*)p)[1];
    const float4 x2 = ((const float4*)p)[2];
    const float4 x3 = ((const float4*)p)[3];
    v16h f;
    f[0]  = (_Float16)x0.x; f[1]  = (_Float16)x0.y; f[2]  = (_Float16)x0.z; f[3]  = (_Float16)x0.w;
    f[4]  = (_Float16)x1.x; f[5]  = (_Float16)x1.y; f[6]  = (_Float16)x1.z; f[7]  = (_Float16)x1.w;
    f[8]  = (_Float16)x2.x; f[9]  = (_Float16)x2.y; f[10] = (_Float16)x2.z; f[11] = (_Float16)x2.w;
    f[12] = (_Float16)x3.x; f[13] = (_Float16)x3.y; f[14] = (_Float16)x3.z; f[15] = (_Float16)x3.w;
    return f;
}

// A-fragment from an LDS row of f16 (row = one M line, K-major).
// A layout (contiguous K): lanes 0-15 K = {kt+0..15}, lanes 16-31 K = {kt+16..31}.
__device__ __forceinline__ v16h load_afrag_lds(const _Float16* __restrict__ row, int kt, int lane)
{
    const int off = kt + ((lane >> 4) << 4);
    const v8h lo = *(const v8h*)(row + off);
    const v8h hi = *(const v8h*)(row + off + 8);
    v16h f;
#pragma unroll
    for (int e = 0; e < 8; ++e) { f[e] = lo[e]; f[e + 8] = hi[e]; }
    return f;
}

// ---------------------------------------------------------------------------
// Mask 1x1-conv GEMM with TDM-staged A tile.
// y[b][co][s] = sum_c dec[b][c][s] * w[co][c] + bias[co]
// grid(3136, 8) = (spatial tiles of 16, batch), block 256 = 8 waves (one N-tile each).
// ---------------------------------------------------------------------------
__global__ void __launch_bounds__(256) mask_conv_kernel(
    const float* __restrict__ dec, const float* __restrict__ w,
    const float* __restrict__ bias, float* __restrict__ y)
{
    __shared__ __align__(16) float    As32[256 * 16];   // TDM dest: [c][m] fp32 (16 KB)
    __shared__ __align__(16) _Float16 At[16][264];      // transposed [m][c] f16, pitch 264

    const int tid  = threadIdx.x;
    const int lane = tid & 31;
    const int wave = tid >> 5;
    const int b    = blockIdx.y;
    const int s0   = blockIdx.x * 16;

    // ---- TDM: DMA the 16(pos) x 256(chan) fp32 tile into LDS (wave 0 issues) ----
    if (wave == 0) {
        const unsigned long long ga =
            (unsigned long long)(const void*)dec +
            (((unsigned long long)b * 256ull * 50176ull + (unsigned long long)s0) << 2);
        const unsigned lds = (unsigned)(unsigned long long)(void*)As32;
        const v4u g0 = { 1u,                                   // count=1 (valid user D#)
                         lds,                                  // lds_addr
                         (unsigned)ga,                         // global_addr[31:0]
                         ((unsigned)(ga >> 32) & 0x01ffffffu)  // global_addr[56:32]
                         | (2u << 30) };                       // type=2 (image)
        const v8i_t g1 = { (int)(2u << 16),          // data_size=4B, wg_mask=0
                           (int)(50176u << 16),      // tensor_dim0[15:0]=50176
                           (int)(256u << 16),        // dim0[31:16]=0, tensor_dim1[15:0]=256
                           (int)(16u << 16),         // dim1[31:16]=0, tile_dim0=16
                           (int)256,                 // tile_dim1=256, tile_dim2=0
                           (int)50176,               // tensor_dim0_stride[31:0]
                           0, 0 };                   // stride0 hi / dim1_stride = 0
        const v4i_t g2 = { 0, 0, 0, 0 };             // dims 2/3 unused (2-D tile)
        const v4i_t g3 = { 0, 0, 0, 0 };
        const v8i_t g4 = { 0, 0, 0, 0, 0, 0, 0, 0 }; // extra group (dims 3/4) unused
        __builtin_amdgcn_tensor_load_to_lds(g0, g1, g2, g3, g4, 0);
        __builtin_amdgcn_s_wait_tensorcnt(0);
    }
    __syncthreads();

    // ---- transpose + f16 convert: At[m][c] = As32[c*16 + m] ----
    {
        const float* src = As32 + tid * 16;   // channel tid, 16 contiguous positions
#pragma unroll
        for (int m = 0; m < 16; ++m) At[m][tid] = (_Float16)src[m];
    }
    __syncthreads();

    // ---- per-wave 16x16 WMMA tile, K = 256 ----
    const int n = (wave << 4) + (lane & 15);
    v8f acc = {};
#pragma unroll
    for (int kt = 0; kt < 256; kt += 32) {
        v16h af = load_afrag_lds(At[lane & 15], kt, lane);
        v16h bf = load_f32x16_as_f16(w + (long)n * 256 + kt + ((lane >> 4) << 4));
        acc = __builtin_amdgcn_wmma_f32_16x16x32_f16(false, af, false, bf,
                                                     (short)0, acc, false, false);
    }

    const float bv = bias[n];
    float* yp = y + ((long)b * 128 + n) * 50176 + s0 + ((lane >> 4) << 3);
    *(float4*)yp       = make_float4(acc[0] + bv, acc[1] + bv, acc[2] + bv, acc[3] + bv);
    *(float4*)(yp + 4) = make_float4(acc[4] + bv, acc[5] + bv, acc[6] + bv, acc[7] + bv);
}

// ---------------------------------------------------------------------------
// Patch embedding GEMM, K split 4 ways for occupancy.
// dlp[kc][m][n] = sum_{k in chunk kc} dec_patch(m,k) * pe_w[n][k]
// M=1568, N=256, K=65536 (k = c*256 + i*16 + j). grid(98, 4), block 256.
// Per 256-K slab: cooperative LDS stage of A (one channel, 16 patches) as f16
// [m][k], then 8 WMMA K-steps; each wave owns 2 N-tiles (16x32 output).
// ---------------------------------------------------------------------------
__global__ void __launch_bounds__(256) patch_embed_kernel(
    const float* __restrict__ dec, const float* __restrict__ pw,
    float* __restrict__ dlp)
{
    __shared__ __align__(16) _Float16 As[16][264];   // [m][k-in-slab], pitch 264

    const int tid  = threadIdx.x;
    const int lane = tid & 31;
    const int wave = tid >> 5;
    const int mt   = blockIdx.x * 16;
    const int kc   = blockIdx.y;                     // K chunk 0..3
    const int ks0  = kc * 16384;

    // staging source: thread -> (patch row m, image row i0) of current channel
    const int ms = mt + (tid >> 4);
    const int bs = ms / 196, psp = ms % 196;
    const int py = psp / 14, px = psp % 14;
    const float* abase = dec + ((long)bs * 256 * 224 + (long)py * 16) * 224 + px * 16;
    const int i0 = tid & 15;

    const int n0 = (wave << 5) + (lane & 15);        // first N column
    v8f acc0 = {}, acc1 = {};

    for (int ks = ks0; ks < ks0 + 16384; ks += 256) {
        // ---- stage one channel (16x16 per patch) of 16 patches as f16 ----
        {
            const int c0 = ks >> 8;
            const float* src = abase + (long)c0 * 50176 + i0 * 224;
            if (c0 + 1 < 256) __builtin_prefetch(src + 50176, 0, 1);
            const float4 x0 = ((const float4*)src)[0];
            const float4 x1 = ((const float4*)src)[1];
            const float4 x2 = ((const float4*)src)[2];
            const float4 x3 = ((const float4*)src)[3];
            v8h h0, h1;
            h0[0] = (_Float16)x0.x; h0[1] = (_Float16)x0.y; h0[2] = (_Float16)x0.z; h0[3] = (_Float16)x0.w;
            h0[4] = (_Float16)x1.x; h0[5] = (_Float16)x1.y; h0[6] = (_Float16)x1.z; h0[7] = (_Float16)x1.w;
            h1[0] = (_Float16)x2.x; h1[1] = (_Float16)x2.y; h1[2] = (_Float16)x2.z; h1[3] = (_Float16)x2.w;
            h1[4] = (_Float16)x3.x; h1[5] = (_Float16)x3.y; h1[6] = (_Float16)x3.z; h1[7] = (_Float16)x3.w;
            _Float16* dst = &As[tid >> 4][i0 * 16];
            *(v8h*)dst       = h0;
            *(v8h*)(dst + 8) = h1;
        }
        __syncthreads();

        // ---- 8 WMMA K-steps over the slab ----
#pragma unroll
        for (int k8 = 0; k8 < 8; ++k8) {
            const int kt = k8 * 32;
            v16h af = load_afrag_lds(As[lane & 15], kt, lane);
            const long kb = (long)ks + kt + ((lane >> 4) << 4);
            v16h b0 = load_f32x16_as_f16(pw + (long)n0 * 65536 + kb);
            v16h b1 = load_f32x16_as_f16(pw + (long)(n0 + 16) * 65536 + kb);
            acc0 = __builtin_amdgcn_wmma_f32_16x16x32_f16(false, af, false, b0,
                                                          (short)0, acc0, false, false);
            acc1 = __builtin_amdgcn_wmma_f32_16x16x32_f16(false, af, false, b1,
                                                          (short)0, acc1, false, false);
        }
        __syncthreads();
    }

    // ---- store partials: dlp[kc][m][n] ----
    float* dp = dlp + (long)kc * 401408;
#pragma unroll
    for (int r = 0; r < 8; ++r) {
        const int m = mt + r + ((lane >> 4) << 3);
        dp[(long)m * 256 + n0]      = acc0[r];
        dp[(long)m * 256 + n0 + 16] = acc1[r];
    }
}

// dl[m][n] = sum_kc dlp[kc][m][n] + pe_b[n];  grid(1568), block 256
__global__ void patch_reduce_kernel(const float* __restrict__ dlp,
                                    const float* __restrict__ pb,
                                    float* __restrict__ dl)
{
    const int i = blockIdx.x * 256 + threadIdx.x;
    dl[i] = dlp[i] + dlp[i + 401408] + dlp[i + 2 * 401408] + dlp[i + 3 * 401408]
          + pb[i & 255];
}

// ---------------------------------------------------------------------------
// Generic small batched GEMM (one wave per 16x16 tile), bounds-guarded.
// Fragment element (rc, k): TRANS=false -> P[rc*ld+k]; TRANS=true -> P[k*ld+rc].
// ---------------------------------------------------------------------------
template<bool TRANS, bool BPAT>
__device__ __forceinline__ v16h load_frag(const float* __restrict__ P, long ld,
                                          int rc, int kt, int Kmax, bool rcok,
                                          int lane)
{
    v16h f;
#pragma unroll
    for (int j = 0; j < 8; ++j) {
        int k = kt + (BPAT ? (((lane >> 4) << 4) + (j << 1))
                           : (((lane >> 4) << 4) + (j << 1)));
        float a0 = 0.0f, a1 = 0.0f;
        if (rcok) {
            if (k     < Kmax) a0 = TRANS ? P[(long)k * ld + rc]       : P[(long)rc * ld + k];
            if (k + 1 < Kmax) a1 = TRANS ? P[(long)(k + 1) * ld + rc] : P[(long)rc * ld + k + 1];
        }
        f[2 * j]     = (_Float16)a0;
        f[2 * j + 1] = (_Float16)a1;
    }
    return f;
}

template<bool TA, bool TB>
__global__ void gemm16_kernel(const float* __restrict__ A, const float* __restrict__ B,
                              float* __restrict__ C, const float* __restrict__ bias,
                              int M, int N, int K,
                              long lda, long ldb, long ldc,
                              long sA, long sB, long sC)
{
    const int lane = threadIdx.x & 31;
    const int wave = threadIdx.x >> 5;
    const int nt = (blockIdx.x * (blockDim.x >> 5) + wave) * 16;
    const int mt = blockIdx.y * 16;
    if (nt >= N || mt >= M) return;                 // wave-uniform exit

    const float* Ab = A + (long)blockIdx.z * sA;
    const float* Bb = B + (long)blockIdx.z * sB;
    float*       Cb = C + (long)blockIdx.z * sC;

    const int  mrow = mt + (lane & 15);
    const int  ncol = nt + (lane & 15);
    const bool mok  = mrow < M;
    const bool nok  = ncol < N;

    v8f acc = {};
    for (int kt = 0; kt < K; kt += 32) {
        v16h af = load_frag<TA, false>(Ab, lda, mrow, kt, K, mok, lane);
        v16h bf = load_frag<TB, true >(Bb, ldb, ncol, kt, K, nok, lane);
        acc = __builtin_amdgcn_wmma_f32_16x16x32_f16(false, af, false, bf,
                                                     (short)0, acc, false, false);
    }

    if (nok) {
        const float bv = bias ? bias[ncol] : 0.0f;
#pragma unroll
        for (int r = 0; r < 8; ++r) {
            int m = mt + r + ((lane >> 4) << 3);
            if (m < M) Cb[(long)m * ldc + ncol] = acc[r] + bv;
        }
    }
}

// ---------------------------------------------------------------------------
// InstanceNorm over whole 128x128 map + row softmax, one block per batch.
// ---------------------------------------------------------------------------
__global__ void innorm_softmax_kernel(float* __restrict__ sim)
{
    float* S = sim + (long)blockIdx.x * 16384;
    __shared__ float rs[256], rq[256], mv[2];
    float s = 0.f, q = 0.f;
    for (int i = threadIdx.x; i < 16384; i += 256) { float v = S[i]; s += v; q += v * v; }
    rs[threadIdx.x] = s; rq[threadIdx.x] = q;
    __syncthreads();
    for (int off = 128; off; off >>= 1) {
        if (threadIdx.x < off) { rs[threadIdx.x] += rs[threadIdx.x + off];
                                 rq[threadIdx.x] += rq[threadIdx.x + off]; }
        __syncthreads();
    }
    if (threadIdx.x == 0) {
        float m = rs[0] * (1.0f / 16384.0f);
        float var = rq[0] * (1.0f / 16384.0f) - m * m;
        mv[0] = m; mv[1] = rsqrtf(var + 1e-5f);
    }
    __syncthreads();
    const float m = mv[0], inv = mv[1];
    const int wave = threadIdx.x >> 5, lane = threadIdx.x & 31;
    for (int row = wave; row < 128; row += 8) {
        float v[4]; float mx = -3.4e38f;
#pragma unroll
        for (int j = 0; j < 4; ++j) {
            v[j] = (S[row * 128 + lane + 32 * j] - m) * inv;
            mx = fmaxf(mx, v[j]);
        }
        for (int off = 16; off; off >>= 1) mx = fmaxf(mx, __shfl_xor(mx, off, 32));
        float sum = 0.f;
#pragma unroll
        for (int j = 0; j < 4; ++j) { v[j] = __expf(v[j] - mx); sum += v[j]; }
        for (int off = 16; off; off >>= 1) sum += __shfl_xor(sum, off, 32);
        const float r = 1.0f / sum;
#pragma unroll
        for (int j = 0; j < 4; ++j) S[row * 128 + lane + 32 * j] = v[j] * r;
    }
}

// ---------------------------------------------------------------------------
// BN statistics (deterministic two-stage reductions)
// ---------------------------------------------------------------------------
__global__ void bn1_partial_kernel(const float* __restrict__ y,
                                   float* __restrict__ ps, float* __restrict__ pq)
{
    const int c = blockIdx.x, b = blockIdx.y;
    const float* Y = y + ((long)b * 128 + c) * 50176;
    float s = 0.f, q = 0.f;
    for (int i = threadIdx.x; i < 50176; i += 256) { float v = Y[i]; s += v; q += v * v; }
    __shared__ float rs[256], rq[256];
    rs[threadIdx.x] = s; rq[threadIdx.x] = q;
    __syncthreads();
    for (int off = 128; off; off >>= 1) {
        if (threadIdx.x < off) { rs[threadIdx.x] += rs[threadIdx.x + off];
                                 rq[threadIdx.x] += rq[threadIdx.x + off]; }
        __syncthreads();
    }
    if (threadIdx.x == 0) { ps[c * 8 + b] = rs[0]; pq[c * 8 + b] = rq[0]; }
}

__global__ void bn1_final_kernel(const float* __restrict__ ps, const float* __restrict__ pq,
                                 float* __restrict__ m1, float* __restrict__ i1)
{
    const int c = threadIdx.x;   // 128 threads
    float s = 0.f, q = 0.f;
    for (int b = 0; b < 8; ++b) { s += ps[c * 8 + b]; q += pq[c * 8 + b]; }
    const float n = 8.0f * 50176.0f;
    const float m = s / n;
    const float var = q / n - m * m;
    m1[c] = m; i1[c] = rsqrtf(var + 1e-5f);
}

__global__ void bn2_stats_kernel(const float* __restrict__ rec14,
                                 float* __restrict__ m2, float* __restrict__ i2)
{
    const int c = blockIdx.x;
    float s = 0.f, q = 0.f;
    for (int i = threadIdx.x; i < 1568; i += 256) {
        float v = rec14[(long)i * 128 + c];
        s += v; q += v * v;
    }
    __shared__ float rs[256], rq[256];
    rs[threadIdx.x] = s; rq[threadIdx.x] = q;
    __syncthreads();
    for (int off = 128; off; off >>= 1) {
        if (threadIdx.x < off) { rs[threadIdx.x] += rs[threadIdx.x + off];
                                 rq[threadIdx.x] += rq[threadIdx.x + off]; }
        __syncthreads();
    }
    if (threadIdx.x == 0) {
        const float n = 1568.0f;
        const float m = rs[0] / n;
        const float var = rq[0] / n - m * m;
        m2[c] = m; i2[c] = rsqrtf(var + 1e-5f);
    }
}

// ---------------------------------------------------------------------------
// Final fusion: out = ReLU(BN2(upsample(rec14))) * ReLU(BN1(y)), float4.
// ---------------------------------------------------------------------------
__global__ void final_kernel(const float* __restrict__ y, const float* __restrict__ rec14,
                             const float* __restrict__ m1, const float* __restrict__ i1,
                             const float* __restrict__ g1, const float* __restrict__ b1,
                             const float* __restrict__ m2, const float* __restrict__ i2,
                             const float* __restrict__ g2, const float* __restrict__ b2,
                             float* __restrict__ out)
{
    const int c = blockIdx.y, b = blockIdx.z;
    const long base = ((long)b * 128 + c) * 50176;
    const int s0 = (blockIdx.x * blockDim.x + threadIdx.x) * 4;

    const float a1 = i1[c] * g1[c], o1 = b1[c] - m1[c] * a1;
    const float a2 = i2[c] * g2[c], o2 = b2[c] - m2[c] * a2;

    const float4 yv = *(const float4*)(y + base + s0);
    const float yj[4] = { yv.x, yv.y, yv.z, yv.w };
    const int h = s0 / 224, w0 = s0 % 224;          // 224 % 4 == 0 -> one row per float4
    const int prow = (h >> 4) * 14;
    const float* R = rec14 + (long)b * 196 * 128 + c;

    float o[4];
#pragma unroll
    for (int j = 0; j < 4; ++j) {
        int p = prow + ((w0 + j) >> 4);
        float rec = fmaxf(R[(long)p * 128] * a2 + o2, 0.0f);
        float msk = fmaxf(yj[j] * a1 + o1, 0.0f);
        o[j] = rec * msk;
    }
    *(float4*)(out + base + s0) = make_float4(o[0], o[1], o[2], o[3]);
}

// ---------------------------------------------------------------------------

extern "C" void kernel_launch(void* const* d_in, const int* in_sizes, int n_in,
                              void* d_out, int out_size, void* d_ws, size_t ws_size,
                              hipStream_t stream)
{
    (void)in_sizes; (void)n_in; (void)out_size; (void)ws_size;

    const float* dec     = (const float*)d_in[0];
    const float* trans   = (const float*)d_in[1];
    const float* pe_w    = (const float*)d_in[2];
    const float* pe_b    = (const float*)d_in[3];
    const float* convm_w = (const float*)d_in[4];
    const float* convm_b = (const float*)d_in[5];
    const float* bn1_g   = (const float*)d_in[6];
    const float* bn1_b   = (const float*)d_in[7];
    const float* wq      = (const float*)d_in[8];
    const float* wk      = (const float*)d_in[9];
    const float* wv      = (const float*)d_in[10];
    const float* wo      = (const float*)d_in[11];
    const float* rec_w   = (const float*)d_in[12];
    const float* rec_b   = (const float*)d_in[13];
    const float* bn2_g   = (const float*)d_in[14];
    const float* bn2_b   = (const float*)d_in[15];
    float* out = (float*)d_out;
    float* ws  = (float*)d_ws;

    // workspace layout (floats)
    size_t off = 0;
    float* y     = ws + off; off += 51380224;   // [8][128][50176] mask conv pre-BN
    float* dlp   = ws + off; off += 1605632;    // [4][1568][256] patch partials
    float* dl    = ws + off; off += 401408;     // [1568][256]
    float* qt    = ws + off; off += 200704;     // [8][196][128]
    float* kk    = ws + off; off += 200704;
    float* vt    = ws + off; off += 200704;
    float* sim   = ws + off; off += 131072;     // [8][128][128]
    float* ot    = ws + off; off += 200704;
    float* oo    = ws + off; off += 200704;
    float* rec14 = ws + off; off += 200704;     // [8*196][128]
    float* ps    = ws + off; off += 1024;
    float* pq    = ws + off; off += 1024;
    float* m1    = ws + off; off += 128;
    float* i1    = ws + off; off += 128;
    float* m2    = ws + off; off += 128;
    float* i2    = ws + off; off += 128;

    // 1) mask pre-BN (TDM-staged A)
    mask_conv_kernel<<<dim3(3136, 8), 256, 0, stream>>>(dec, convm_w, convm_b, y);

    // 2) patch embedding (K split 4 ways) + reduce
    patch_embed_kernel<<<dim3(98, 4), 256, 0, stream>>>(dec, pe_w, dlp);
    patch_reduce_kernel<<<1568, 256, 0, stream>>>(dlp, pe_b, dl);

    // 3) qt[n][c] = sum_d dl[n][d] * wq[d][c]       (B stored [K][N] -> TB=true)
    gemm16_kernel<false, true><<<dim3(1, 13, 8), 256, 0, stream>>>(
        dl, wq, qt, nullptr, 196, 128, 256, 256, 128, 128,
        (long)196 * 256, 0, (long)196 * 128);

    // 4) kk[n][c] = trans[n][e] * wk[e][c]
    gemm16_kernel<false, true><<<dim3(1, 13, 8), 256, 0, stream>>>(
        trans, wk, kk, nullptr, 196, 128, 960, 960, 128, 128,
        (long)196 * 960, 0, (long)196 * 128);

    // 5) vt[n][c] = trans[n][e] * wv[e][c]
    gemm16_kernel<false, true><<<dim3(1, 13, 8), 256, 0, stream>>>(
        trans, wv, vt, nullptr, 196, 128, 960, 960, 128, 128,
        (long)196 * 960, 0, (long)196 * 128);

    // 6) sim[c][d] = sum_n qt[n][c] * kk[n][d]   (A trans, B stored [K][N])
    gemm16_kernel<true, true><<<dim3(1, 8, 8), 256, 0, stream>>>(
        qt, kk, sim, nullptr, 128, 128, 196, 128, 128, 128,
        (long)196 * 128, (long)196 * 128, (long)128 * 128);

    // 7) instance-norm + softmax on sim
    innorm_softmax_kernel<<<8, 256, 0, stream>>>(sim);

    // 8) ot[n][c] = sum_d vt[n][d] * sim[c][d]   (B stored [N][K] -> TB=false)
    gemm16_kernel<false, false><<<dim3(1, 13, 8), 256, 0, stream>>>(
        vt, sim, ot, nullptr, 196, 128, 128, 128, 128, 128,
        (long)196 * 128, (long)128 * 128, (long)196 * 128);

    // 9) oo[n][d] = sum_c ot[n][c] * wo[c][d]    (B stored [K][N] -> TB=true)
    gemm16_kernel<false, true><<<dim3(1, 13, 8), 256, 0, stream>>>(
        ot, wo, oo, nullptr, 196, 128, 128, 128, 128, 128,
        (long)196 * 128, 0, (long)196 * 128);

    // 10) rec14[p][co] = sum_d oo[p][d] * rec_w[co][d] + rec_b[co]  (TB=false)
    gemm16_kernel<false, false><<<dim3(1, 13, 8), 256, 0, stream>>>(
        oo, rec_w, rec14, rec_b, 196, 128, 128, 128, 128, 128,
        (long)196 * 128, 0, (long)196 * 128);

    // 11-13) BN statistics
    bn1_partial_kernel<<<dim3(128, 8), 256, 0, stream>>>(y, ps, pq);
    bn1_final_kernel<<<1, 128, 0, stream>>>(ps, pq, m1, i1);
    bn2_stats_kernel<<<128, 256, 0, stream>>>(rec14, m2, i2);

    // 14) fused upsample + BN2 + ReLU, BN1 + ReLU, multiply
    final_kernel<<<dim3(49, 128, 8), 256, 0, stream>>>(
        y, rec14, m1, i1, bn1_g, bn1_b, m2, i2, bn2_g, bn2_b, out);
}
